// PointNetFeaturePropagation_2439541424197
// MI455X (gfx1250) — matrix operand order repaired and checked by reading, not compiled, not run on hardware
//
#include <hip/hip_runtime.h>
#include <cstdint>

// ---------------------------------------------------------------------------
// PointNet Feature Propagation on MI455X (gfx1250, wave32, WMMA).
// GEMMs: v_wmma_f32_16x16x32_bf16 with split-bf16 (hi+lo) operands (~f32
// accuracy at bf16 matrix-pipe rate). B operand staged to LDS with CDNA5
// GLOBAL_LOAD_ASYNC_TO_LDS_B128 (ASYNCcnt) double buffering.
// ---------------------------------------------------------------------------

typedef __attribute__((ext_vector_type(16))) __bf16 v16bf;
typedef __attribute__((ext_vector_type(8)))  __bf16 v8bf;
typedef __attribute__((ext_vector_type(8)))  float  v8f;

#define DEV __device__ __forceinline__

static constexpr int BB = 16, NN = 4096, SS = 1024, DD = 128;
static constexpr int PP = BB * NN;          // 65536 GEMM columns
static constexpr int K0 = 512, M0 = 256;    // layer 0
static constexpr int K1 = 256, M1 = 128;    // layer 1

// LDS staging geometry for GEMM B tiles (256 cols x 32 halfs, hi+lo, x2 bufs)
static constexpr int COL_H  = 40;                   // padded col stride (halfs), 80B: 16B aligned, bank-conflict-free
static constexpr int REG_H  = 256 * COL_H;          // one matrix region (halfs) = 10240
static constexpr int BUF_H  = 2 * REG_H;            // hi+lo per buffer (halfs)  = 20480
static constexpr int LDS_BYTES = 2 * BUF_H * 2;     // double buffered           = 81920

// ---- workspace layout (bytes) ---------------------------------------------
static constexpr size_t OFF_IDX = 0x000000;                 // int   [P*3]
static constexpr size_t OFF_WGT = 0x0C0000;                 // float [P*3]
static constexpr size_t OFF_W0H = 0x180000;                 // bf16  [256*512]
static constexpr size_t OFF_W0L = 0x1C0000;
static constexpr size_t OFF_W1H = 0x200000;                 // bf16  [128*256]
static constexpr size_t OFF_W1L = 0x210000;
static constexpr size_t OFF_SC0 = 0x220000;                 // float [256]
static constexpr size_t OFF_SH0 = 0x220400;
static constexpr size_t OFF_SC1 = 0x221000;                 // float [128]
static constexpr size_t OFF_SH1 = 0x221200;
static constexpr size_t OFF_XHI = 0x400000;                 // bf16 [P][512]  64MiB
static constexpr size_t OFF_XLO = OFF_XHI + 0x4000000;      // bf16 [P][512]  64MiB
static constexpr size_t OFF_Y0  = OFF_XLO + 0x4000000;      // f32  [256][P]  64MiB
// reuse after gemm0 consumed X:
static constexpr size_t OFF_HHI = OFF_XHI;                  // bf16 [P][256]  32MiB
static constexpr size_t OFF_HLO = OFF_XHI + 0x2000000;      // bf16 [P][256]  32MiB
static constexpr size_t OFF_Y1  = OFF_XLO;                  // f32  [128][P]  32MiB

// ---------------------------------------------------------------------------
DEV v8f wmma_bf16(v16bf a, v16bf b, v8f c) {
  return __builtin_amdgcn_wmma_f32_16x16x32_bf16(
      false, a, false, b, (short)0, c, false, false);
}

// A-fragment (16x32 bf16): lanes 0-15 -> M=lane, 16 consecutive K at sel*16
DEV v16bf load_a_frag(const __bf16* __restrict__ row, int sel) {
  union { v16bf v; v8bf h[2]; } u;
  u.h[0] = *(const v8bf*)(row + sel * 16);
  u.h[1] = *(const v8bf*)(row + sel * 16 + 8);
  return u.v;
}

// B-fragment (32x16 bf16): lane holds N=lane&15, 16 consecutive K at sel*16
DEV v16bf load_b_frag(const __bf16* __restrict__ p) {
  union { v16bf v; v8bf h[2]; } u;
  u.h[0] = *(const v8bf*)(p);
  u.h[1] = *(const v8bf*)(p + 8);
  return u.v;
}

DEV void split_store(__bf16* __restrict__ h, __bf16* __restrict__ l, int i, float v) {
  __bf16 hi = (__bf16)v;
  h[i] = hi;
  l[i] = (__bf16)(v - (float)hi);
}

// CDNA5 async VMEM -> LDS copy (ASYNCcnt). lds_off = LDS byte address.
DEV void async_b128(unsigned lds_off, const __bf16* g) {
  asm volatile("global_load_async_to_lds_b128 %0, %1, off"
               :: "v"(lds_off), "v"(g)
               : "memory");
}

// Stage one 256-col x 32-half B tile (hi+lo) into LDS buffer at lds_base.
// 8 async b128 per thread (2048 chunks of 16B).
DEV void stage_b_tile(const __bf16* __restrict__ Bhi, const __bf16* __restrict__ Blo,
                      int n0, int K, int kb, unsigned lds_base, int tid) {
  const __bf16* gh = Bhi + (size_t)n0 * K + kb;
  const __bf16* gl = Blo + (size_t)n0 * K + kb;
#pragma unroll
  for (int j = 0; j < 4; ++j) {
    int c = tid + 256 * j;                 // chunk id
    int col = c >> 2, part = c & 3;        // 4 x 16B per column
    unsigned lo_off = (unsigned)(col * (COL_H * 2) + part * 16);
    async_b128(lds_base + lo_off, gh + (size_t)col * K + part * 8);
  }
#pragma unroll
  for (int j = 0; j < 4; ++j) {
    int c = tid + 256 * j;
    int col = c >> 2, part = c & 3;
    unsigned lo_off = (unsigned)(REG_H * 2 + col * (COL_H * 2) + part * 16);
    async_b128(lds_base + lo_off, gl + (size_t)col * K + part * 8);
  }
}

// ---------------------------------------------------------------------------
// Kernel 1: 3-NN + inverse-distance weights.  grid(N/256, B), block 256.
// ---------------------------------------------------------------------------
__global__ void knn3_kernel(const float* __restrict__ xyz1,
                            const float* __restrict__ xyz2,
                            int* __restrict__ nn_idx, float* __restrict__ nn_w) {
  __shared__ float sx[SS], sy[SS], sz[SS], sn[SS];
  const int b = blockIdx.y;
  const float* x2 = xyz2 + (size_t)b * 3 * SS;
  for (int i = threadIdx.x; i < SS; i += blockDim.x) {
    float a = x2[i], c = x2[SS + i], d = x2[2 * SS + i];
    sx[i] = a; sy[i] = c; sz[i] = d; sn[i] = a * a + c * c + d * d;
  }
  __syncthreads();
  const int n = blockIdx.x * blockDim.x + threadIdx.x;
  const float* x1 = xyz1 + (size_t)b * 3 * NN;
  const float px = x1[n], py = x1[NN + n], pz = x1[2 * NN + n];
  const float pn = px * px + py * py + pz * pz;
  float d0 = 3.4e38f, d1 = 3.4e38f, d2 = 3.4e38f;
  int i0 = 0, i1 = 0, i2 = 0;
  for (int s = 0; s < SS; ++s) {
    float dot = px * sx[s] + py * sy[s] + pz * sz[s];
    float dd = pn + sn[s] - 2.0f * dot;           // matches reference formula
    bool c0 = dd < d0, c1 = dd < d1, c2 = dd < d2;
    d2 = c1 ? d1 : (c2 ? dd : d2);  i2 = c1 ? i1 : (c2 ? s : i2);
    d1 = c0 ? d0 : (c1 ? dd : d1);  i1 = c0 ? i0 : (c1 ? s : i1);
    d0 = c0 ? dd : d0;              i0 = c0 ? s  : i0;
  }
  float r0 = 1.0f / (d0 + 1e-8f), r1 = 1.0f / (d1 + 1e-8f), r2 = 1.0f / (d2 + 1e-8f);
  float rs = 1.0f / (r0 + r1 + r2);
  size_t p3 = ((size_t)b * NN + n) * 3;
  nn_idx[p3 + 0] = i0; nn_idx[p3 + 1] = i1; nn_idx[p3 + 2] = i2;
  nn_w[p3 + 0] = r0 * rs; nn_w[p3 + 1] = r1 * rs; nn_w[p3 + 2] = r2 * rs;
}

// ---------------------------------------------------------------------------
// Kernel 2: build X [P][512] in split bf16, n-major (K contiguous).
// ---------------------------------------------------------------------------
__global__ void build_x_kernel(const float* __restrict__ points1,
                               const float* __restrict__ points2,
                               const int* __restrict__ nn_idx,
                               const float* __restrict__ nn_w,
                               __bf16* __restrict__ Xhi, __bf16* __restrict__ Xlo) {
  const int n = blockIdx.x * blockDim.x + threadIdx.x;
  const int b = blockIdx.y;
  const size_t p = (size_t)b * NN + n;
  const float* p1 = points1 + (size_t)b * DD * NN + n;   // stride NN per channel
  const float* p2 = points2 + (size_t)b * DD * SS;       // [D][S]
  const int i0 = nn_idx[p * 3 + 0], i1 = nn_idx[p * 3 + 1], i2 = nn_idx[p * 3 + 2];
  const float w0 = nn_w[p * 3 + 0], w1 = nn_w[p * 3 + 1], w2 = nn_w[p * 3 + 2];
  __bf16* xh = Xhi + p * K0;
  __bf16* xl = Xlo + p * K0;
  for (int c = 0; c < DD; ++c)
    split_store(xh, xl, c, p1[(size_t)c * NN]);
  for (int d = 0; d < DD; ++d) {
    const float* col = p2 + (size_t)d * SS;
    split_store(xh, xl, 128 + d, w0 * col[i0]);
    split_store(xh, xl, 256 + d, w1 * col[i1]);
    split_store(xh, xl, 384 + d, w2 * col[i2]);
  }
}

// ---------------------------------------------------------------------------
// Kernel 3: split f32 weights into bf16 hi/lo.
// ---------------------------------------------------------------------------
__global__ void split_w_kernel(const float* __restrict__ w,
                               __bf16* __restrict__ hi, __bf16* __restrict__ lo, int n) {
  int i = blockIdx.x * blockDim.x + threadIdx.x;
  if (i < n) split_store(hi, lo, i, w[i]);
}

// ---------------------------------------------------------------------------
// Kernel 4: GEMM  Y[M][P] = A[M][K] * X[P][K]^T + bias, split-bf16 operands.
// Block = 256 threads = 8 waves; block tile 32(M) x 256(N); wave tile 16x64.
// B tile async-staged to LDS (double buffered).  grid(P/256, M/32).
// ---------------------------------------------------------------------------
__global__ void gemm_kernel(const __bf16* __restrict__ Ahi, const __bf16* __restrict__ Alo,
                            const float* __restrict__ bias,
                            const __bf16* __restrict__ Bhi, const __bf16* __restrict__ Blo,
                            float* __restrict__ Y, int M, int K) {
  extern __shared__ __bf16 smem[];                       // LDS_BYTES
  const unsigned smem_base = (unsigned)(uintptr_t)(void*)smem;  // LDS byte address (low 32 bits)
  const int tid  = threadIdx.x;
  const int lane = tid & 31;
  const int wave = tid >> 5;
  const int m_off = blockIdx.y * 32 + (wave & 1) * 16;
  const int n0    = blockIdx.x * 256;
  const int wn    = wave >> 1;                           // 0..3 -> 64-col strip
  const int lm    = lane & 15;
  const int sel   = lane >> 4;

  v8f zero = {0.f, 0.f, 0.f, 0.f, 0.f, 0.f, 0.f, 0.f};
  v8f acc[4] = {zero, zero, zero, zero};

  const __bf16* arow_h = Ahi + (size_t)(m_off + lm) * K;
  const __bf16* arow_l = Alo + (size_t)(m_off + lm) * K;
  // loop-invariant LDS read base: column (wn*64 + s2*16 + lm), K-offset sel*16
  const __bf16* lds_rd = smem + (size_t)(wn * 64 + lm) * COL_H + sel * 16;

  stage_b_tile(Bhi, Blo, n0, K, 0, smem_base, tid);

  for (int kb = 0; kb < K; kb += 32) {
    const int buf = (kb >> 5) & 1;
    if (kb + 32 < K) {
      stage_b_tile(Bhi, Blo, n0, K, kb + 32, smem_base + (unsigned)((buf ^ 1) * BUF_H * 2), tid);
      asm volatile("s_wait_asynccnt 8" ::: "memory");    // previous batch done; new one in flight
    } else {
      asm volatile("s_wait_asynccnt 0" ::: "memory");
    }
    __syncthreads();                                     // tile for 'buf' visible to all waves

    v16bf ah = load_a_frag(arow_h + kb, sel);
    v16bf al = load_a_frag(arow_l + kb, sel);
    const __bf16* bp = lds_rd + (size_t)buf * BUF_H;
#pragma unroll
    for (int s2 = 0; s2 < 4; ++s2) {
      v16bf bh = load_b_frag(bp + s2 * 16 * COL_H);
      v16bf bl = load_b_frag(bp + REG_H + s2 * 16 * COL_H);
      // (ah+al)*(bh+bl) ~= ah*bh + ah*bl + al*bh   (al*bl below f32 ulp)
      acc[s2] = wmma_bf16(ah, bh, acc[s2]);
      acc[s2] = wmma_bf16(ah, bl, acc[s2]);
      acc[s2] = wmma_bf16(al, bh, acc[s2]);
    }
    __syncthreads();                                     // protect buffer reuse (2 buffers)
  }

  // C/D layout: lanes0-15 -> M=r, N=lane; lanes16-31 -> M=8+r, N=lane-16.
#pragma unroll
  for (int s2 = 0; s2 < 4; ++s2) {
    int n = n0 + wn * 64 + s2 * 16 + lm;
#pragma unroll
    for (int r = 0; r < 8; ++r) {
      int m = m_off + sel * 8 + r;
      Y[(size_t)m * PP + n] = acc[s2][r] + bias[m];
    }
  }
}

// ---------------------------------------------------------------------------
// Kernel 5: BN stats per channel -> scale/shift.  grid(M), block 256.
// ---------------------------------------------------------------------------
__global__ void bn_stats_kernel(const float* __restrict__ Y,
                                const float* __restrict__ gamma,
                                const float* __restrict__ beta,
                                float* __restrict__ scale, float* __restrict__ shift) {
  const int m = blockIdx.x;
  const float* row = Y + (size_t)m * PP;
  float s = 0.f, q = 0.f;
  for (int i = threadIdx.x; i < PP; i += blockDim.x) {
    float v = row[i];
    s += v; q += v * v;
  }
  __shared__ float ss[256], sq[256];
  ss[threadIdx.x] = s; sq[threadIdx.x] = q;
  __syncthreads();
  for (int o = 128; o > 0; o >>= 1) {
    if (threadIdx.x < o) { ss[threadIdx.x] += ss[threadIdx.x + o]; sq[threadIdx.x] += sq[threadIdx.x + o]; }
    __syncthreads();
  }
  if (threadIdx.x == 0) {
    float mean = ss[0] / (float)PP;
    float var  = sq[0] / (float)PP - mean * mean;
    float sc = gamma[m] * rsqrtf(var + 1e-5f);
    scale[m] = sc;
    shift[m] = beta[m] - mean * sc;
  }
}

// ---------------------------------------------------------------------------
// Kernel 6: BN+ReLU, re-split to bf16 [P][M] for next GEMM. grid(P/256), blk 256.
// ---------------------------------------------------------------------------
__global__ void bn_apply_split_kernel(const float* __restrict__ Y,
                                      const float* __restrict__ scale,
                                      const float* __restrict__ shift,
                                      __bf16* __restrict__ Hhi, __bf16* __restrict__ Hlo,
                                      int M) {
  const int p = blockIdx.x * blockDim.x + threadIdx.x;
  __bf16* hh = Hhi + (size_t)p * M;
  __bf16* hl = Hlo + (size_t)p * M;
  for (int m = 0; m < M; ++m) {
    float v = fmaxf(Y[(size_t)m * PP + p] * scale[m] + shift[m], 0.f);
    split_store(hh, hl, m, v);
  }
}

// ---------------------------------------------------------------------------
// Kernel 7: final BN+ReLU -> out [B,128,N] f32.  grid(M1*P/256), blk 256.
// ---------------------------------------------------------------------------
__global__ void bn_apply_out_kernel(const float* __restrict__ Y,
                                    const float* __restrict__ scale,
                                    const float* __restrict__ shift,
                                    float* __restrict__ out) {
  const int t = blockIdx.x * blockDim.x + threadIdx.x;   // over M1 * P
  const int m = t >> 16;           // P = 65536
  const int p = t & 65535;
  const int b = p >> 12;           // N = 4096
  const int n = p & 4095;
  float v = fmaxf(Y[(size_t)m * PP + p] * scale[m] + shift[m], 0.f);
  out[((size_t)b * M1 + m) * NN + n] = v;
}

// ---------------------------------------------------------------------------
extern "C" void kernel_launch(void* const* d_in, const int* in_sizes, int n_in,
                              void* d_out, int out_size, void* d_ws, size_t ws_size,
                              hipStream_t stream) {
  (void)in_sizes; (void)n_in; (void)out_size; (void)ws_size;
  const float* xyz1    = (const float*)d_in[0];
  const float* xyz2    = (const float*)d_in[1];
  const float* points1 = (const float*)d_in[2];
  const float* points2 = (const float*)d_in[3];
  const float* w0      = (const float*)d_in[4];
  const float* b0      = (const float*)d_in[5];
  const float* g0      = (const float*)d_in[6];
  const float* be0     = (const float*)d_in[7];
  const float* w1      = (const float*)d_in[8];
  const float* b1      = (const float*)d_in[9];
  const float* g1      = (const float*)d_in[10];
  const float* be1     = (const float*)d_in[11];
  float* out = (float*)d_out;

  char* ws = (char*)d_ws;
  int*    nn_idx = (int*)   (ws + OFF_IDX);
  float*  nn_w   = (float*) (ws + OFF_WGT);
  __bf16* w0hi   = (__bf16*)(ws + OFF_W0H);
  __bf16* w0lo   = (__bf16*)(ws + OFF_W0L);
  __bf16* w1hi   = (__bf16*)(ws + OFF_W1H);
  __bf16* w1lo   = (__bf16*)(ws + OFF_W1L);
  float*  sc0    = (float*) (ws + OFF_SC0);
  float*  sh0    = (float*) (ws + OFF_SH0);
  float*  sc1    = (float*) (ws + OFF_SC1);
  float*  sh1    = (float*) (ws + OFF_SH1);
  __bf16* Xhi    = (__bf16*)(ws + OFF_XHI);
  __bf16* Xlo    = (__bf16*)(ws + OFF_XLO);
  float*  Y0     = (float*) (ws + OFF_Y0);
  __bf16* Hhi    = (__bf16*)(ws + OFF_HHI);   // overwrites Xhi (dead after gemm0)
  __bf16* Hlo    = (__bf16*)(ws + OFF_HLO);
  float*  Y1     = (float*) (ws + OFF_Y1);    // overwrites Xlo (dead after gemm0)

  dim3 blk(256);

  knn3_kernel<<<dim3(NN / 256, BB), blk, 0, stream>>>(xyz1, xyz2, nn_idx, nn_w);
  build_x_kernel<<<dim3(NN / 256, BB), blk, 0, stream>>>(points1, points2, nn_idx, nn_w, Xhi, Xlo);
  split_w_kernel<<<dim3((M0 * K0 + 255) / 256), blk, 0, stream>>>(w0, w0hi, w0lo, M0 * K0);
  split_w_kernel<<<dim3((M1 * K1 + 255) / 256), blk, 0, stream>>>(w1, w1hi, w1lo, M1 * K1);

  gemm_kernel<<<dim3(PP / 256, M0 / 32), blk, LDS_BYTES, stream>>>(w0hi, w0lo, b0, Xhi, Xlo, Y0, M0, K0);
  bn_stats_kernel<<<dim3(M0), blk, 0, stream>>>(Y0, g0, be0, sc0, sh0);
  bn_apply_split_kernel<<<dim3(PP / 256), blk, 0, stream>>>(Y0, sc0, sh0, Hhi, Hlo, M0);

  gemm_kernel<<<dim3(PP / 256, M1 / 32), blk, LDS_BYTES, stream>>>(w1hi, w1lo, b1, Hhi, Hlo, Y1, M1, K1);
  bn_stats_kernel<<<dim3(M1), blk, 0, stream>>>(Y1, g1, be1, sc1, sh1);
  bn_apply_out_kernel<<<dim3(M1 * PP / 256), blk, 0, stream>>>(Y1, sc1, sh1, out);
}
